// MC_Mamba_32134945308971
// MI455X (gfx1250) — matrix operand order repaired and checked by reading, not compiled, not run on hardware
//
#include <hip/hip_runtime.h>
#include <hip/hip_bf16.h>

// ---------------- problem constants ----------------
#define BATCH   8
#define DMODEL  192
#define DINNER  384      // 2*DMODEL
#define DSTATE  16
#define DTRANK  12
#define HH      48
#define WWI     48
#define LLEN    2304     // HH*WWI
#define NROWS   18432    // BATCH*LLEN
#define NPROJ   44       // DTRANK + 2*DSTATE
#define NPROJP  48       // padded to multiple of 16
#define EPSLN   1e-5f

typedef __attribute__((ext_vector_type(16))) __bf16 v16bf;
typedef __attribute__((ext_vector_type(8)))  __bf16 v8bf;
typedef __attribute__((ext_vector_type(8)))  float  v8f;

__device__ __forceinline__ __bf16 to_bf16(float f) {
  union { float f; unsigned u; } v; v.f = f;
  unsigned r = v.u + 0x7FFFu + ((v.u >> 16) & 1u);   // round-to-nearest-even
  unsigned short h = (unsigned short)(r >> 16);
  __bf16 o; __builtin_memcpy(&o, &h, 2);
  return o;
}

__device__ __forceinline__ float silu(float v) { return v / (1.0f + __expf(-v)); }

// ---------------- generic f32 -> bf16 convert ----------------
__global__ void cvt_f32_bf16_kernel(const float* __restrict__ src,
                                    __bf16* __restrict__ dst, int n) {
  for (int i = blockIdx.x * blockDim.x + threadIdx.x; i < n;
       i += gridDim.x * blockDim.x)
    dst[i] = to_bf16(src[i]);
}

__global__ void zero_bf16_kernel(__bf16* __restrict__ dst, int n) {
  for (int i = blockIdx.x * blockDim.x + threadIdx.x; i < n;
       i += gridDim.x * blockDim.x)
    dst[i] = to_bf16(0.0f);
}

// ---------------- NCHW f32 -> (B,L,C) bf16 ----------------
__global__ void pack_x_kernel(const float* __restrict__ x,
                              __bf16* __restrict__ xbf) {
  const int total = BATCH * DMODEL * LLEN;
  for (int i = blockIdx.x * blockDim.x + threadIdx.x; i < total;
       i += gridDim.x * blockDim.x) {
    int l = i % LLEN;
    int c = (i / LLEN) % DMODEL;
    int b = i / (LLEN * DMODEL);
    xbf[((size_t)b * LLEN + l) * DMODEL + c] = to_bf16(x[i]);
  }
}

// ---------------- WMMA bf16 GEMM:  C[M,N] = A[M,K] * W[N,K]^T ----------------
// A row-major (M,K) bf16 ; W row-major (N,K) bf16 ; C row-major (M,N) f32.
// One wave computes a 16(M) x (NV*16)(N) strip, K stepped by 32.
// M%16==0, K%32==0, N%(NV*16)==0 required (true for all call sites).
// NV is a compile-time constant so acc[] stays in fixed VGPRs (no v_movrel).
template <int NV>
__global__ void gemm_bf16_wmma_kernel(const __bf16* __restrict__ A,
                                      const __bf16* __restrict__ W,
                                      float* __restrict__ Cmat,
                                      int M, int N, int K) {
  const int lane   = threadIdx.x & 31;
  const int waveId = blockIdx.x * (blockDim.x >> 5) + (threadIdx.x >> 5);
  const int nstrips = N / (NV * 16);
  const int mt = waveId / nstrips;
  const int ns = waveId % nstrips;
  if (mt * 16 >= M) return;
  const int m0 = mt * 16;
  const int r  = lane & 15;       // row (A) / column (B) within tile
  const int hf = lane >> 4;       // which K-half this lane owns

  // Hoisted per-lane fragment pointers (bumped by +32 elements per K step).
  const __bf16* pA = A + (size_t)(m0 + r) * K + hf * 8;
  const __bf16* pB[NV];
#pragma unroll
  for (int j = 0; j < NV; ++j)
    pB[j] = W + (size_t)(ns * (NV * 16) + j * 16 + r) * K + hf * 8;

  v8f acc[NV];
#pragma unroll
  for (int j = 0; j < NV; ++j) acc[j] = (v8f)(0.0f);

#pragma unroll 2
  for (int k0 = 0; k0 < K; k0 += 32) {
    // A fragment: lane r holds K = {k0+hf*8 .. +7} and {k0+16+hf*8 .. +7}
    v8bf alo = *(const v8bf*)(pA);
    v8bf ahi = *(const v8bf*)(pA + 16);
    v16bf afrag = __builtin_shufflevector(alo, ahi,
        0, 1, 2, 3, 4, 5, 6, 7, 8, 9, 10, 11, 12, 13, 14, 15);
    __builtin_prefetch(pA + 64, 0, 3);   // -> global_prefetch_b8 (near caches)
    pA += 32;

#pragma unroll
    for (int j = 0; j < NV; ++j) {
      // B fragment: lane (hf,r) holds W[n0+r, k0+hf*8 .. +7] and W[n0+r, k0+16+hf*8 .. +7] (a B column)
      v8bf blo = *(const v8bf*)(pB[j]);
      v8bf bhi = *(const v8bf*)(pB[j] + 16);
      v16bf bfrag = __builtin_shufflevector(blo, bhi,
          0, 1, 2, 3, 4, 5, 6, 7, 8, 9, 10, 11, 12, 13, 14, 15);
      pB[j] += 32;

      acc[j] = __builtin_amdgcn_wmma_f32_16x16x32_bf16(
          false, afrag, false, bfrag, (short)0, acc[j], false, false);
    }
  }

  // D layout: VGPR q -> (M = m0 + hf*8 + q, N = n0 + (lane&15))
  const int mOff = hf * 8;
#pragma unroll
  for (int j = 0; j < NV; ++j) {
    const int n0 = ns * (NV * 16) + j * 16;
    float* outp = Cmat + (size_t)(m0 + mOff) * N + n0 + r;
#pragma unroll
    for (int q = 0; q < 8; ++q) outp[(size_t)q * N] = acc[j][q];
  }
}

// ------------- depthwise 3x3 conv (SAME) + bias + SiLU on xz[..., :384] -------------
// writes u (f32) and xact (bf16), both (B,L,DINNER)
__global__ void conv_dw_silu_kernel(const float* __restrict__ xz,
                                    const float* __restrict__ conv_w,
                                    const float* __restrict__ conv_b,
                                    float* __restrict__ u,
                                    __bf16* __restrict__ xact) {
  const int total = BATCH * LLEN * DINNER;
  for (int i = blockIdx.x * blockDim.x + threadIdx.x; i < total;
       i += gridDim.x * blockDim.x) {
    int d = i % DINNER;
    int l = (i / DINNER) % LLEN;
    int b = i / (DINNER * LLEN);
    int h = l / WWI, w = l % WWI;
    float acc = conv_b[d];
    #pragma unroll
    for (int dh = -1; dh <= 1; ++dh) {
      int hh = h + dh;
      if (hh < 0 || hh >= HH) continue;
      #pragma unroll
      for (int dw = -1; dw <= 1; ++dw) {
        int ww = w + dw;
        if (ww < 0 || ww >= WWI) continue;
        float xv = xz[((size_t)b * LLEN + hh * WWI + ww) * (2 * DINNER) + d];
        acc += conv_w[d * 9 + (dh + 1) * 3 + (dw + 1)] * xv;
      }
    }
    float s = silu(acc);
    u[i] = s;
    xact[i] = to_bf16(s);
  }
}

// ------------- delta = softplus(dt_low @ dt_proj_w^T + dt_proj_b) -------------
__global__ void dt_softplus_kernel(const float* __restrict__ xp,
                                   const float* __restrict__ dt_proj_w,
                                   const float* __restrict__ dt_proj_b,
                                   float* __restrict__ delta) {
  const int total = NROWS * DINNER;
  for (int i = blockIdx.x * blockDim.x + threadIdx.x; i < total;
       i += gridDim.x * blockDim.x) {
    int d = i % DINNER;
    int row = i / DINNER;
    const float* xr = xp + (size_t)row * NPROJP;
    float acc = dt_proj_b[d];
    #pragma unroll
    for (int rr = 0; rr < DTRANK; ++rr) acc += xr[rr] * dt_proj_w[d * DTRANK + rr];
    delta[i] = (acc > 20.0f) ? acc : log1pf(__expf(acc));
  }
}

// ------------- selective scan: 16 lanes = 16 states per (b,d); 2 pairs per wave -------------
__global__ void scan_kernel(const float* __restrict__ delta,
                            const float* __restrict__ u,
                            const float* __restrict__ xp,
                            const float* __restrict__ A_log,
                            const float* __restrict__ D_param,
                            float* __restrict__ y) {
  const int lane = threadIdx.x & 31;
  const int wave = (blockIdx.x * blockDim.x + threadIdx.x) >> 5;
  const int pair = wave * 2 + (lane >> 4);          // (b,d) index
  if (pair >= BATCH * DINNER) return;
  const int b = pair / DINNER;
  const int d = pair % DINNER;
  const int s = lane & 15;

  const float As = -__expf(A_log[d * DSTATE + s]);
  const float Dd = D_param[d];

  const float* dlt = delta + (size_t)b * LLEN * DINNER + d;
  const float* uu  = u     + (size_t)b * LLEN * DINNER + d;
  const float* xpb = xp    + (size_t)b * LLEN * NPROJP;
  float*       yo  = y     + (size_t)b * LLEN * DINNER + d;

  float h = 0.0f;
  for (int l = 0; l < LLEN; ++l) {
    float dv = dlt[(size_t)l * DINNER];
    float uv = uu[(size_t)l * DINNER];
    float Bv = xpb[l * NPROJP + DTRANK + s];
    float Cv = xpb[l * NPROJP + DTRANK + DSTATE + s];
    h = __expf(dv * As) * h + (dv * uv) * Bv;
    float p = h * Cv;
    p += __shfl_xor(p, 1);
    p += __shfl_xor(p, 2);
    p += __shfl_xor(p, 4);
    p += __shfl_xor(p, 8);
    if (s == 0) yo[(size_t)l * DINNER] = p + uv * Dd;
  }
}

// ------------- LayerNorm(384) * SiLU(z) -> bf16 ; one wave per row -------------
__global__ void ln_mul_silu_kernel(const float* __restrict__ y,
                                   const float* __restrict__ xz,
                                   const float* __restrict__ ln_g,
                                   const float* __restrict__ ln_b,
                                   __bf16* __restrict__ ybf) {
  const int row  = blockIdx.x * (blockDim.x >> 5) + (threadIdx.x >> 5);
  const int lane = threadIdx.x & 31;
  if (row >= NROWS) return;
  const float* yr = y + (size_t)row * DINNER;
  float s = 0.f, s2 = 0.f;
  #pragma unroll
  for (int i = 0; i < DINNER / 32; ++i) {
    float v = yr[lane + i * 32];
    s += v; s2 += v * v;
  }
  #pragma unroll
  for (int m = 16; m >= 1; m >>= 1) {
    s  += __shfl_xor(s,  m);
    s2 += __shfl_xor(s2, m);
  }
  const float mu   = s / (float)DINNER;
  const float var  = s2 / (float)DINNER - mu * mu;
  const float rstd = rsqrtf(var + EPSLN);
  #pragma unroll
  for (int i = 0; i < DINNER / 32; ++i) {
    int d = lane + i * 32;
    float v = (yr[d] - mu) * rstd * ln_g[d] + ln_b[d];
    float z = xz[(size_t)row * (2 * DINNER) + DINNER + d];
    ybf[(size_t)row * DINNER + d] = to_bf16(v * silu(z));
  }
}

// ------------- instance-norm stats over L per (b,c) ; one wave each -------------
__global__ void inst_stats_kernel(const float* __restrict__ fused,
                                  float* __restrict__ mean,
                                  float* __restrict__ rstd) {
  const int idx  = blockIdx.x * (blockDim.x >> 5) + (threadIdx.x >> 5);
  const int lane = threadIdx.x & 31;
  if (idx >= BATCH * DMODEL) return;
  const int b = idx / DMODEL;
  const int c = idx % DMODEL;
  float s = 0.f, s2 = 0.f;
  for (int l = lane; l < LLEN; l += 32) {
    float v = fused[((size_t)b * LLEN + l) * DMODEL + c];
    s += v; s2 += v * v;
  }
  #pragma unroll
  for (int m = 16; m >= 1; m >>= 1) {
    s  += __shfl_xor(s,  m);
    s2 += __shfl_xor(s2, m);
  }
  if (lane == 0) {
    float mu  = s / (float)LLEN;
    float var = s2 / (float)LLEN - mu * mu;
    mean[idx] = mu;
    rstd[idx] = rsqrtf(var + EPSLN);
  }
}

// ------------- residual + normalize, write NCHW f32 -------------
// (fuse_b cancels exactly inside instance norm, so it is omitted)
__global__ void final_kernel(const float* __restrict__ x,
                             const float* __restrict__ fused,
                             const float* __restrict__ mean,
                             const float* __restrict__ rstd,
                             float* __restrict__ out) {
  const int total = BATCH * DMODEL * LLEN;
  for (int i = blockIdx.x * blockDim.x + threadIdx.x; i < total;
       i += gridDim.x * blockDim.x) {
    int l = i % LLEN;
    int c = (i / LLEN) % DMODEL;
    int b = i / (LLEN * DMODEL);
    int idx = b * DMODEL + c;
    float f = fused[((size_t)b * LLEN + l) * DMODEL + c];
    out[i] = x[i] + (f - mean[idx]) * rstd[idx];
  }
}

// =============================== host launcher ===============================
extern "C" void kernel_launch(void* const* d_in, const int* in_sizes, int n_in,
                              void* d_out, int out_size, void* d_ws, size_t ws_size,
                              hipStream_t stream) {
  (void)in_sizes; (void)n_in; (void)out_size; (void)ws_size;

  const float* x          = (const float*)d_in[0];
  const float* in_proj_w  = (const float*)d_in[1];
  const float* conv_w     = (const float*)d_in[2];
  const float* conv_b     = (const float*)d_in[3];
  const float* x_proj_w   = (const float*)d_in[4];
  const float* dt_proj_w  = (const float*)d_in[5];
  const float* dt_proj_b  = (const float*)d_in[6];
  const float* A_log      = (const float*)d_in[7];
  const float* D_param    = (const float*)d_in[8];
  const float* ln_g       = (const float*)d_in[9];
  const float* ln_b       = (const float*)d_in[10];
  const float* out_proj_w = (const float*)d_in[11];
  const float* fuse_w     = (const float*)d_in[12];
  // d_in[13] = fuse_b : cancels in instance norm
  float* out = (float*)d_out;

  // ---- workspace carve-out (256B aligned chunks) ----
  size_t off = 0;
  auto take = [&](size_t bytes) -> void* {
    void* p = (void*)((char*)d_ws + off);
    off += (bytes + 255) & ~(size_t)255;
    return p;
  };
  __bf16* Xbf     = (__bf16*)take((size_t)NROWS * DMODEL * 2);
  __bf16* Winbf   = (__bf16*)take((size_t)(2 * DINNER) * DMODEL * 2);
  __bf16* Wxpbf   = (__bf16*)take((size_t)NPROJP * DINNER * 2);
  __bf16* Woutbf  = (__bf16*)take((size_t)DMODEL * DINNER * 2);
  __bf16* Wfusebf = (__bf16*)take((size_t)DMODEL * DMODEL * 2);
  float*  xz      = (float*)take((size_t)NROWS * (2 * DINNER) * 4);
  float*  u       = (float*)take((size_t)NROWS * DINNER * 4);
  __bf16* xactbf  = (__bf16*)take((size_t)NROWS * DINNER * 2);
  float*  xp      = (float*)take((size_t)NROWS * NPROJP * 4);
  float*  delta   = (float*)take((size_t)NROWS * DINNER * 4);
  float*  y       = (float*)take((size_t)NROWS * DINNER * 4);
  __bf16* ybf     = (__bf16*)take((size_t)NROWS * DINNER * 2);
  float*  proj    = (float*)take((size_t)NROWS * DMODEL * 4);
  __bf16* projbf  = (__bf16*)take((size_t)NROWS * DMODEL * 2);
  float*  fused   = (float*)take((size_t)NROWS * DMODEL * 4);
  float*  meanv   = (float*)take((size_t)BATCH * DMODEL * 4);
  float*  rstdv   = (float*)take((size_t)BATCH * DMODEL * 4);

  const int T = 256;
  auto nb = [](int n, int t) { return (n + t - 1) / t; };

  // GEMM dispatch: strip width 64 (NV=4) when N%64==0, else 48 (NV=3, N=48).
  auto launch_gemm = [&](const __bf16* A, const __bf16* Wt, float* C,
                         int M, int N, int K) {
    if (N % 64 == 0) {
      int waves  = (M / 16) * (N / 64);
      int blocks = (waves + 7) / 8;
      gemm_bf16_wmma_kernel<4><<<blocks, T, 0, stream>>>(A, Wt, C, M, N, K);
    } else {
      int waves  = (M / 16) * (N / 48);
      int blocks = (waves + 7) / 8;
      gemm_bf16_wmma_kernel<3><<<blocks, T, 0, stream>>>(A, Wt, C, M, N, K);
    }
  };

  // ---- weight packing ----
  cvt_f32_bf16_kernel<<<nb(2 * DINNER * DMODEL, T), T, 0, stream>>>(in_proj_w, Winbf, 2 * DINNER * DMODEL);
  zero_bf16_kernel<<<nb(NPROJP * DINNER, T), T, 0, stream>>>(Wxpbf, NPROJP * DINNER);
  cvt_f32_bf16_kernel<<<nb(NPROJ * DINNER, T), T, 0, stream>>>(x_proj_w, Wxpbf, NPROJ * DINNER);
  cvt_f32_bf16_kernel<<<nb(DMODEL * DINNER, T), T, 0, stream>>>(out_proj_w, Woutbf, DMODEL * DINNER);
  cvt_f32_bf16_kernel<<<nb(DMODEL * DMODEL, T), T, 0, stream>>>(fuse_w, Wfusebf, DMODEL * DMODEL);

  // ---- 1) pack x to (B,L,C) bf16 ----
  pack_x_kernel<<<nb(BATCH * DMODEL * LLEN, T), T, 0, stream>>>(x, Xbf);

  // ---- 2) in_proj GEMM: xz = Xbf(18432x192) * Winbf(768x192)^T ----
  launch_gemm(Xbf, Winbf, xz, NROWS, 2 * DINNER, DMODEL);

  // ---- 3) depthwise conv + SiLU ----
  conv_dw_silu_kernel<<<nb(BATCH * LLEN * DINNER, T), T, 0, stream>>>(
      xz, conv_w, conv_b, u, xactbf);

  // ---- 4) x_proj GEMM: xp = xact(18432x384) * Wxp(48x384)^T ----
  launch_gemm(xactbf, Wxpbf, xp, NROWS, NPROJP, DINNER);

  // ---- 5) dt projection + softplus ----
  dt_softplus_kernel<<<nb(NROWS * DINNER, T), T, 0, stream>>>(
      xp, dt_proj_w, dt_proj_b, delta);

  // ---- 6) selective scan ----
  {
    int pairs = BATCH * DINNER;          // 3072
    int waves = (pairs + 1) / 2;         // 1536
    int blocks = (waves + 7) / 8;        // 192
    scan_kernel<<<blocks, T, 0, stream>>>(delta, u, xp, A_log, D_param, y);
  }

  // ---- 7) LayerNorm * SiLU(z) -> bf16 ----
  ln_mul_silu_kernel<<<nb(NROWS, 8), T, 0, stream>>>(y, xz, ln_g, ln_b, ybf);

  // ---- 8) out_proj GEMM: proj = ybf(18432x384) * Wout(192x384)^T ----
  launch_gemm(ybf, Woutbf, proj, NROWS, DMODEL, DINNER);

  // ---- 9) fuse 1x1 GEMM: fused = projbf(18432x192) * Wfuse(192x192)^T ----
  cvt_f32_bf16_kernel<<<nb(NROWS * DMODEL, T), T, 0, stream>>>(proj, projbf, NROWS * DMODEL);
  launch_gemm(projbf, Wfusebf, fused, NROWS, DMODEL, DMODEL);

  // ---- 10) instance norm + residual ----
  inst_stats_kernel<<<nb(BATCH * DMODEL, 8), T, 0, stream>>>(fused, meanv, rstdv);
  final_kernel<<<nb(BATCH * DMODEL * LLEN, T), T, 0, stream>>>(x, fused, meanv, rstdv, out);
}
